// GATHeteroGNN_5686536699972
// MI455X (gfx1250) — matrix operand order repaired and checked by reading, not compiled, not run on hardware
//
#include <hip/hip_runtime.h>
#include <cmath>

#define NN 100000          // nodes per type
#define NE 500000          // edges per relation
#define HC 128             // H*C
#define NH 4               // heads
#define CW 32              // channels per head
#define SLOPE 0.2f
#define EPSF 1e-16f

typedef __attribute__((ext_vector_type(16))) __bf16 v16bf;
typedef __attribute__((ext_vector_type(8)))  float  v8f;

__device__ __forceinline__ unsigned short f2bf(float f) {
  unsigned int u = __float_as_uint(f);
  u += 0x7FFFu + ((u >> 16) & 1u);          // round-to-nearest-even
  return (unsigned short)(u >> 16);
}
__device__ __forceinline__ float bf2f(unsigned short h) {
  return __uint_as_float(((unsigned int)h) << 16);
}

__device__ __forceinline__ void atomicMaxF(float* addr, float v) {
  if (v >= 0.0f) atomicMax((int*)addr, __float_as_int(v));
  else           atomicMin((unsigned int*)addr, __float_as_uint(v));
}

// ---------------------------------------------------------------------------
// Fold weights.
//  - Wswz: lin_src (zero-padded to 128x128) converted to bf16 and laid out in
//    WMMA B-fragment order: index ((kt*8 + wave)*32 + lane)*16 + i, where
//    element i of lane L maps to K = kt*32 + (i<8 ? (L<16?0:8)+i : 16+(L<16?0:8)+i-8), N = wave*16+L%16.
//  - w_dst[F,4]  = lin_dst contracted with att_dst   (h_dst never materialized)
//  - w_edge[8,4] = lin_edge contracted with att_edge (h_edge never materialized)
// ---------------------------------------------------------------------------
__global__ void prep_rel(const float* __restrict__ lin_src,
                         const float* __restrict__ lin_dst,
                         const float* __restrict__ lin_edge,
                         const float* __restrict__ att_dst,
                         const float* __restrict__ att_edge,
                         unsigned short* __restrict__ wswz,
                         float* __restrict__ wdst,
                         float* __restrict__ wedge, int F) {
  int stride = gridDim.x * blockDim.x;
  int tid = blockIdx.x * blockDim.x + threadIdx.x;
  for (int o = tid; o < HC * HC; o += stride) {
    int i    = o & 15;
    int lane = (o >> 4) & 31;
    int w    = (o >> 9) & 7;
    int kt   = o >> 12;
    int k = kt * 32 + ((i < 8) ? (((lane < 16) ? 0 : 8) + i) : (16 + ((lane < 16) ? 0 : 8) + (i - 8)));
    int n = w * 16 + (lane & 15);
    float v = (k < F) ? lin_src[k * HC + n] : 0.0f;
    wswz[o] = f2bf(v);
  }
  for (int i = tid; i < F * NH; i += stride) {
    int f = i >> 2, h = i & 3;
    float s = 0.0f;
    for (int c = 0; c < CW; ++c) s += lin_dst[f * HC + h * CW + c] * att_dst[h * CW + c];
    wdst[i] = s;
  }
  for (int i = tid; i < 8 * NH; i += stride) {
    int k = i >> 2, h = i & 3;
    float s = 0.0f;
    for (int c = 0; c < CW; ++c) s += lin_edge[k * HC + h * CW + c] * att_edge[h * CW + c];
    wedge[i] = s;
  }
}

// ---------------------------------------------------------------------------
// H[N,128] (bf16) = X[N,F] @ W[F,128] via v_wmma_f32_16x16x32_bf16.
// Block = 256 threads (8 waves) = one 16-row tile; wave w owns columns
// [16w,16w+16). A staged in LDS in fragment order (2x ds_load_b128/lane/kt);
// B read from pre-swizzled global (2x global_load_b128/lane/kt). No branches
// or EXEC divergence in the MMA loop. Epilogue repacks D through LDS into
// coalesced global_store_b128.
// ---------------------------------------------------------------------------
template <int KT>
__global__ void __launch_bounds__(256)
gemm_h_wmma(const void* __restrict__ Xv, int srcIsF32,
            const unsigned short* __restrict__ Wswz,
            unsigned short* __restrict__ Hout, int F) {
  __shared__ __align__(16) unsigned short sA[4 * 32 * 16];  // [kt][lane][i]
  __shared__ __align__(16) unsigned short sD[16 * HC];
  const int mBase = blockIdx.x * 16;
  const int tid = threadIdx.x;

  // stage & convert X tile directly into A-fragment order
  for (int idx = tid; idx < KT * 32 * 16; idx += 256) {
    int i    = idx & 15;
    int lane = (idx >> 4) & 31;
    int kt   = idx >> 9;
    int half8 = (lane < 16) ? 0 : 8;
    int ka = (i < 8) ? (half8 + i) : (16 + half8 + (i - 8));
    int k = kt * 32 + ka;
    int row = mBase + (lane & 15);
    float v = 0.0f;
    if (k < F)
      v = srcIsF32 ? ((const float*)Xv)[(size_t)row * F + k]
                   : bf2f(((const unsigned short*)Xv)[(size_t)row * F + k]);
    sA[idx] = f2bf(v);
  }
  __syncthreads();

  const int wave = tid >> 5;
  const int lane = tid & 31;
  union Frag { v16bf v; uint4 q[2]; } fa, fb;
  v8f acc = {};
  const uint4* sAq = (const uint4*)sA;
  const uint4* wq  = (const uint4*)Wswz;
#pragma unroll
  for (int kt = 0; kt < KT; ++kt) {
    int ai = (kt * 32 + lane) * 2;                 // 16 ushorts = 2x uint4
    fa.q[0] = sAq[ai];
    fa.q[1] = sAq[ai + 1];
    int bi = ((kt * 8 + wave) * 32 + lane) * 2;
    fb.q[0] = wq[bi];
    fb.q[1] = wq[bi + 1];
    acc = __builtin_amdgcn_wmma_f32_16x16x32_bf16(false, fa.v, false, fb.v,
                                                  (short)0, acc, false, false);
  }

  // D layout: VGPR r -> M = r (lanes 0-15) or 8+r (lanes 16-31), N = lane%16
  const int half8d = (lane < 16) ? 0 : 8;
  const int nCol = wave * 16 + (lane & 15);
#pragma unroll
  for (int r = 0; r < 8; ++r) sD[(half8d + r) * HC + nCol] = f2bf(acc[r]);
  __syncthreads();

  // cooperative coalesced store: 256 threads x 16B = 4KB tile
  uint4* Hq = (uint4*)(Hout + (size_t)mBase * HC);
  Hq[tid] = ((const uint4*)sD)[tid];
}

// a_src[n,h] = sum_c H[n,h*32+c] * att[h*32+c]
__global__ void node_logits(const unsigned short* __restrict__ H,
                            const float* __restrict__ att,
                            float* __restrict__ out, int N) {
  int n = blockIdx.x * blockDim.x + threadIdx.x;
  if (n >= N) return;
  const uint2* hr = (const uint2*)(H + (size_t)n * HC);   // 4 bf16 per uint2
#pragma unroll
  for (int h = 0; h < NH; ++h) {
    float s = 0.0f;
    for (int c4 = 0; c4 < CW / 4; ++c4) {
      uint2 p = hr[h * (CW / 4) + c4];
      int c = c4 * 4;
      s += bf2f((unsigned short)(p.x & 0xFFFFu)) * att[h * CW + c + 0];
      s += bf2f((unsigned short)(p.x >> 16))     * att[h * CW + c + 1];
      s += bf2f((unsigned short)(p.y & 0xFFFFu)) * att[h * CW + c + 2];
      s += bf2f((unsigned short)(p.y >> 16))     * att[h * CW + c + 3];
    }
    out[n * NH + h] = s;
  }
}

// a_dst[n,h] = sum_f X[n,f] * wdst[f,h]
__global__ void dst_logits(const void* __restrict__ Xv, int srcIsF32,
                           const float* __restrict__ wdst,
                           float* __restrict__ out, int N, int F) {
  int n = blockIdx.x * blockDim.x + threadIdx.x;
  if (n >= N) return;
  float s[NH] = {0.f, 0.f, 0.f, 0.f};
  for (int f = 0; f < F; ++f) {
    float x = srcIsF32 ? ((const float*)Xv)[(size_t)n * F + f]
                       : bf2f(((const unsigned short*)Xv)[(size_t)n * F + f]);
#pragma unroll
    for (int h = 0; h < NH; ++h) s[h] += x * wdst[f * NH + h];
  }
#pragma unroll
  for (int h = 0; h < NH; ++h) out[n * NH + h] = s[h];
}

__global__ void fill_f32(float* __restrict__ p, float v, int n) {
  int i = blockIdx.x * blockDim.x + threadIdx.x;
  if (i < n) p[i] = v;
}

// pass 1: raw logit + leaky relu, atomic segment max
__global__ void edge_alpha_max(const int* __restrict__ src, const int* __restrict__ dst,
                               const float* __restrict__ ea,
                               const float* __restrict__ asrc, const float* __restrict__ adst,
                               const float* __restrict__ wedge,
                               float* __restrict__ raw, float* __restrict__ amax, int E) {
  int e = blockIdx.x * blockDim.x + threadIdx.x;
  if (e >= E) return;
  int s = src[e], d = dst[e];
  float lo[NH];
#pragma unroll
  for (int h = 0; h < NH; ++h) lo[h] = asrc[s * NH + h] + adst[d * NH + h];
#pragma unroll
  for (int k = 0; k < 8; ++k) {
    float v = ea[e * 8 + k];
#pragma unroll
    for (int h = 0; h < NH; ++h) lo[h] += v * wedge[k * NH + h];
  }
#pragma unroll
  for (int h = 0; h < NH; ++h) {
    float a = lo[h];
    a = (a > 0.0f) ? a : SLOPE * a;
    raw[e * NH + h] = a;
    atomicMaxF(&amax[d * NH + h], a);
  }
}

// pass 2: exp(raw - max), atomic denom sum; raw buffer now holds exp values
__global__ void edge_exp_sum(const int* __restrict__ dst, float* __restrict__ raw,
                             const float* __restrict__ amax, float* __restrict__ den, int E) {
  int e = blockIdx.x * blockDim.x + threadIdx.x;
  if (e >= E) return;
  int d = dst[e];
#pragma unroll
  for (int h = 0; h < NH; ++h) {
    float m = amax[d * NH + h];
    if (m < -3.0e38f) m = 0.0f;               // isfinite guard
    float a = __expf(raw[e * NH + h] - m);
    raw[e * NH + h] = a;
    atomicAdd(&den[d * NH + h], a);
  }
}

// pass 3: wave-per-edge scatter; lane owns 4 consecutive channels -> one 8B
// vector gather per lane, then 4 f32 atomic adds into the L2-resident dst row.
__global__ void edge_scatter(const int* __restrict__ src, const int* __restrict__ dst,
                             const float* __restrict__ alpha, const float* __restrict__ den,
                             const unsigned short* __restrict__ H, float* __restrict__ acc,
                             int E) {
  int gid = blockIdx.x * blockDim.x + threadIdx.x;
  int e = gid >> 5;
  int lane = gid & 31;
  if (e >= E) return;
  int s = src[e], d = dst[e];
  int h  = lane >> 3;          // 8 lanes per head
  int c0 = lane * 4;           // 4 consecutive channels per lane
  float w = alpha[e * NH + h] / (den[d * NH + h] + EPSF);
  uint2 p = *(const uint2*)(H + (size_t)s * HC + c0);
  float* ap = acc + (size_t)d * HC + c0;
  atomicAdd(ap + 0, bf2f((unsigned short)(p.x & 0xFFFFu)) * w);
  atomicAdd(ap + 1, bf2f((unsigned short)(p.x >> 16)) * w);
  atomicAdd(ap + 2, bf2f((unsigned short)(p.y & 0xFFFFu)) * w);
  atomicAdd(ap + 3, bf2f((unsigned short)(p.y >> 16)) * w);
}

// bias + relu -> bf16 activations for next layer
__global__ void node_epilogue(const float* __restrict__ acc, const float* __restrict__ bias,
                              unsigned short* __restrict__ Xn, int total) {
  int i = blockIdx.x * blockDim.x + threadIdx.x;
  if (i >= total) return;
  float v = acc[i] + bias[i & (HC - 1)];
  Xn[i] = f2bf(v > 0.0f ? v : 0.0f);
}

// out[n,:2] = a[n,:] @ lin_w + lin_b
__global__ void final_linear(const unsigned short* __restrict__ A,
                             const float* __restrict__ W, const float* __restrict__ b,
                             float* __restrict__ out, int N) {
  int n = blockIdx.x * blockDim.x + threadIdx.x;
  if (n >= N) return;
  float s0 = b[0], s1 = b[1];
  const uint2* ar = (const uint2*)(A + (size_t)n * HC);
  for (int k4 = 0; k4 < HC / 4; ++k4) {
    uint2 p = ar[k4];
    int k = k4 * 4;
    s0 += bf2f((unsigned short)(p.x & 0xFFFFu)) * W[(k + 0) * 2];
    s1 += bf2f((unsigned short)(p.x & 0xFFFFu)) * W[(k + 0) * 2 + 1];
    s0 += bf2f((unsigned short)(p.x >> 16)) * W[(k + 1) * 2];
    s1 += bf2f((unsigned short)(p.x >> 16)) * W[(k + 1) * 2 + 1];
    s0 += bf2f((unsigned short)(p.y & 0xFFFFu)) * W[(k + 2) * 2];
    s1 += bf2f((unsigned short)(p.y & 0xFFFFu)) * W[(k + 2) * 2 + 1];
    s0 += bf2f((unsigned short)(p.y >> 16)) * W[(k + 3) * 2];
    s1 += bf2f((unsigned short)(p.y >> 16)) * W[(k + 3) * 2 + 1];
  }
  out[n * 2 + 0] = s0;
  out[n * 2 + 1] = s1;
}

// ---------------------------------------------------------------------------
static inline size_t alignup(size_t x) { return (x + 255) & ~(size_t)255; }

extern "C" void kernel_launch(void* const* d_in, const int* in_sizes, int n_in,
                              void* d_out, int out_size, void* d_ws, size_t ws_size,
                              hipStream_t stream) {
  (void)in_sizes; (void)n_in; (void)out_size; (void)ws_size;

  const float* x_addr = (const float*)d_in[0];
  const float* x_tx   = (const float*)d_in[1];
  const int*   ei_in  = (const int*)d_in[2];   // [2,E] addr->tx
  const int*   ei_out = (const int*)d_in[3];   // [2,E] tx->addr
  const float* ea_in  = (const float*)d_in[4];
  const float* ea_out = (const float*)d_in[5];
  // params flattened in setup_inputs insertion order:
  // input[0..2] then output[0..2], each: lin_src,lin_dst,lin_edge,att_src,att_dst,att_edge,bias
  auto P = [&](int rel, int l, int k) -> const float* {   // rel: 0=input,1=output
    return (const float*)d_in[6 + rel * 21 + l * 7 + k];
  };
  const float* lin_w = (const float*)d_in[48];
  const float* lin_b = (const float*)d_in[49];

  char* ws = (char*)d_ws;
  size_t off = 0;
  auto grab = [&](size_t bytes) { char* p = ws + off; off += alignup(bytes); return p; };

  const size_t HB = (size_t)NN * HC * sizeof(unsigned short);
  const size_t FB = (size_t)NN * HC * sizeof(float);
  const size_t LB = (size_t)NN * NH * sizeof(float);
  const size_t EB = (size_t)NE * NH * sizeof(float);

  unsigned short* A_cur = (unsigned short*)grab(HB);
  unsigned short* T_cur = (unsigned short*)grab(HB);
  unsigned short* H_in  = (unsigned short*)grab(HB);
  unsigned short* H_out = (unsigned short*)grab(HB);
  float* ACC_T = (float*)grab(FB);
  float* ACC_A = (float*)grab(FB);
  float* asrc_i = (float*)grab(LB);
  float* adst_i = (float*)grab(LB);
  float* asrc_o = (float*)grab(LB);
  float* adst_o = (float*)grab(LB);
  float* amax_t = (float*)grab(LB);
  float* den_t  = (float*)grab(LB);
  float* amax_a = (float*)grab(LB);
  float* den_a  = (float*)grab(LB);
  float* raw_i  = (float*)grab(EB);
  float* raw_o  = (float*)grab(EB);
  unsigned short* wswz[6]; float* wdst[6]; float* wedge[6];
  for (int r = 0; r < 6; ++r) {
    wswz[r]  = (unsigned short*)grab((size_t)HC * HC * sizeof(unsigned short));
    wdst[r]  = (float*)grab((size_t)HC * NH * sizeof(float));
    wedge[r] = (float*)grab((size_t)8 * NH * sizeof(float));
  }

  // ---- weight folding (6 relation-layers) ----
  for (int l = 0; l < 3; ++l) {
    int F = (l == 0) ? 16 : HC;
    for (int rel = 0; rel < 2; ++rel) {
      int r = rel * 3 + l;
      prep_rel<<<64, 256, 0, stream>>>(P(rel, l, 0), P(rel, l, 1), P(rel, l, 2),
                                       P(rel, l, 4), P(rel, l, 5),
                                       wswz[r], wdst[r], wedge[r], F);
    }
  }

  const int nodeGrid = (NN + 255) / 256;
  const int edgeGrid = (NE + 255) / 256;
  const int scatGrid = (NE + 7) / 8;           // 8 edges per 256-thread block
  const int accGrid  = (NN * HC + 255) / 256;
  const int segGrid  = (NN * NH + 255) / 256;
  const int mTiles   = NN / 16;                // 6250, exact
  const float NEG_INF = -INFINITY;

  const void* Xa = (const void*)x_addr;
  const void* Xt = (const void*)x_tx;
  int isF32 = 1, F = 16;

  for (int l = 0; l < 3; ++l) {
    int ri = 0 * 3 + l, ro = 1 * 3 + l;
    // node transforms (WMMA) — h_src for each relation
    if (l == 0) {
      gemm_h_wmma<1><<<mTiles, 256, 0, stream>>>(Xa, isF32, wswz[ri], H_in, F);
      gemm_h_wmma<1><<<mTiles, 256, 0, stream>>>(Xt, isF32, wswz[ro], H_out, F);
    } else {
      gemm_h_wmma<4><<<mTiles, 256, 0, stream>>>(Xa, isF32, wswz[ri], H_in, F);
      gemm_h_wmma<4><<<mTiles, 256, 0, stream>>>(Xt, isF32, wswz[ro], H_out, F);
    }
    // attention logits
    node_logits<<<nodeGrid, 256, 0, stream>>>(H_in,  P(0, l, 3), asrc_i, NN);
    node_logits<<<nodeGrid, 256, 0, stream>>>(H_out, P(1, l, 3), asrc_o, NN);
    dst_logits<<<nodeGrid, 256, 0, stream>>>(Xt, isF32, wdst[ri], adst_i, NN, F); // dst of input rel = tx
    dst_logits<<<nodeGrid, 256, 0, stream>>>(Xa, isF32, wdst[ro], adst_o, NN, F); // dst of output rel = addr
    // init segment state + accumulators
    fill_f32<<<segGrid, 256, 0, stream>>>(amax_t, NEG_INF, NN * NH);
    fill_f32<<<segGrid, 256, 0, stream>>>(den_t,  0.0f,    NN * NH);
    fill_f32<<<segGrid, 256, 0, stream>>>(amax_a, NEG_INF, NN * NH);
    fill_f32<<<segGrid, 256, 0, stream>>>(den_a,  0.0f,    NN * NH);
    fill_f32<<<accGrid, 256, 0, stream>>>(ACC_T, 0.0f, NN * HC);
    fill_f32<<<accGrid, 256, 0, stream>>>(ACC_A, 0.0f, NN * HC);
    // input relation: addr -> tx
    edge_alpha_max<<<edgeGrid, 256, 0, stream>>>(ei_in, ei_in + NE, ea_in,
                                                 asrc_i, adst_i, wedge[ri], raw_i, amax_t, NE);
    edge_exp_sum<<<edgeGrid, 256, 0, stream>>>(ei_in + NE, raw_i, amax_t, den_t, NE);
    edge_scatter<<<scatGrid, 256, 0, stream>>>(ei_in, ei_in + NE,
                                               raw_i, den_t, H_in, ACC_T, NE);
    // output relation: tx -> addr
    edge_alpha_max<<<edgeGrid, 256, 0, stream>>>(ei_out, ei_out + NE, ea_out,
                                                 asrc_o, adst_o, wedge[ro], raw_o, amax_a, NE);
    edge_exp_sum<<<edgeGrid, 256, 0, stream>>>(ei_out + NE, raw_o, amax_a, den_a, NE);
    edge_scatter<<<scatGrid, 256, 0, stream>>>(ei_out, ei_out + NE,
                                               raw_o, den_a, H_out, ACC_A, NE);
    // bias + relu -> next-layer activations (bf16)
    node_epilogue<<<accGrid, 256, 0, stream>>>(ACC_T, P(0, l, 6), T_cur, NN * HC);
    node_epilogue<<<accGrid, 256, 0, stream>>>(ACC_A, P(1, l, 6), A_cur, NN * HC);
    Xa = (const void*)A_cur;
    Xt = (const void*)T_cur;
    isF32 = 0;
    F = HC;
  }

  final_linear<<<nodeGrid, 256, 0, stream>>>(A_cur, lin_w, lin_b, (float*)d_out, NN);
}
